// GraphEncoder_70987219468418
// MI455X (gfx1250) — matrix operand order repaired and checked by reading, not compiled, not run on hardware
//
#include <hip/hip_runtime.h>

#define N_NODES   100000
#define N_EDGES   1600000
#define F_IN      128
#define HID       128
#define LAT       64
#define NUM_LAYERS 3
#define NUM_GRAPHS 512

typedef __attribute__((ext_vector_type(16))) __bf16 v16bf;
typedef __attribute__((ext_vector_type(8)))  __bf16 v8bf;
typedef __attribute__((ext_vector_type(8)))  float  v8f;

// Guaranteed native f32 atomic add (L2-resident RMW), tracked by STOREcnt.
__device__ __forceinline__ void gatomic_add_f32(float* p, float v) {
    asm volatile("global_atomic_add_f32 %0, %1, off" :: "v"(p), "v"(v) : "memory");
}

// ---- A-operand loaders (16x32 bf16 WMMA layout) --------------------------
// VGPR0..3 hold K=ka..ka+7, VGPR4..7 hold K=ka+16..ka+23 for this half-wave.
__device__ __forceinline__ void load_a(const float* __restrict__ p, int ka, v16bf& a) {
#pragma unroll
    for (int e = 0; e < 8; ++e) {          // f32 source: b128 loads + cvt_pk_bf16
        a[e]     = (__bf16)p[ka + e];
        a[e + 8] = (__bf16)p[ka + 16 + e];
    }
}
__device__ __forceinline__ void load_a(const __bf16* __restrict__ p, int ka, v16bf& a) {
    v8bf lo = *(const v8bf*)(p + ka);      // bf16 source: two 16B loads, no cvt
    v8bf hh = *(const v8bf*)(p + ka + 16);
#pragma unroll
    for (int e = 0; e < 8; ++e) { a[e] = lo[e]; a[e + 8] = hh[e]; }
}

// ---- Weight pre-pack: f32 [128,ncols] -> bf16 in exact B-operand layout ---
// Packed element ((t*4+kc)*32 + lane)*16 + e  holds  W[(kc*32 + (e>>3)*16 + (lane>>4)*8 + (e&7))*ncols + t*16 + (lane&15)]
// so each wave reads its B chunk as one contiguous 32B-per-lane vector.
__global__ __launch_bounds__(256) void pack_w_kernel(
    const float* __restrict__ W, __bf16* __restrict__ out, int ncols) {
    const int idx  = blockIdx.x * 256 + threadIdx.x;
    const int e    = idx & 15;
    const int lane = (idx >> 4) & 31;
    const int kc   = (idx >> 9) & 3;
    const int t    = idx >> 11;
    const int col  = t * 16 + (lane & 15);
    const int k    = kc * 32 + (e >> 3) * 16 + (lane >> 4) * 8 + (e & 7);
    out[idx] = (__bf16)W[(size_t)k * ncols + col];
}

// ---- Node GEMM: out(bf16) = relu(X @ W + bias) [+ res(bf16)] -------------
// X:[rows,128] (f32 or bf16). One 16-row block, 8 waves = 128 cols.
// K=128 as 4 x v_wmma_f32_16x16x32_bf16, f32 accumulate.
template <typename AT, bool RES>
__global__ __launch_bounds__(256) void node_gemm_kernel(
    const AT* __restrict__ X, const __bf16* __restrict__ Wp,
    const float* __restrict__ bias, const __bf16* __restrict__ res,
    __bf16* __restrict__ out) {
    const int lane = threadIdx.x & 31;
    const int wave = threadIdx.x >> 5;
    const int hi   = lane >> 4;
    const int l15  = lane & 15;
    const int rowBase = blockIdx.x * 16;
    const int row  = rowBase + l15;
    const int col  = wave * 16 + l15;

    const AT* __restrict__ xrow = X + (size_t)row * HID;
    const v16bf* __restrict__ bp = (const v16bf*)Wp + (size_t)(wave * 4) * 32 + lane;
    v8f acc = {};
#pragma unroll
    for (int kc = 0; kc < 4; ++kc) {
        v16bf a;
        load_a(xrow, kc * 32 + hi * 8, a);
        v16bf b = bp[kc * 32];             // pre-packed B chunk: 2 x b128 per lane
        acc = __builtin_amdgcn_wmma_f32_16x16x32_bf16(false, a, false, b,
                                                      (short)0, acc, false, false);
    }
    const float bv = bias[col];
#pragma unroll
    for (int r = 0; r < 8; ++r) {
        const int m = rowBase + r + hi * 8;
        float v = acc[r] + bv;
        v = v > 0.f ? v : 0.f;
        if (RES) v += (float)res[(size_t)m * HID + col];
        out[(size_t)m * HID + col] = (__bf16)v;
    }
}

// ---- Edge scatter: agg[dst] += msg[src] ----------------------------------
// One wave per edge; bf16 gather (256B/edge, half the f32 traffic), widen to
// f32, native f32 atomics into L2-resident agg.
__global__ __launch_bounds__(256) void scatter_kernel(
    const __bf16* __restrict__ msg, const int* __restrict__ src,
    const int* __restrict__ dst, float* __restrict__ agg) {
    const int e = blockIdx.x * 8 + (threadIdx.x >> 5);
    if (e >= N_EDGES) return;
    const int lane = threadIdx.x & 31;
    const int s = src[e];
    const int d = dst[e];
    const uint2 pv = *(const uint2*)(msg + (size_t)s * HID + lane * 4);
    float* p = agg + (size_t)d * HID + lane * 4;
    gatomic_add_f32(p + 0, __uint_as_float(pv.x << 16));
    gatomic_add_f32(p + 1, __uint_as_float(pv.x & 0xffff0000u));
    gatomic_add_f32(p + 2, __uint_as_float(pv.y << 16));
    gatomic_add_f32(p + 3, __uint_as_float(pv.y & 0xffff0000u));
}

// ---- Graph sum-pool: g[batch[node]] += h[node] ---------------------------
__global__ __launch_bounds__(128) void pool_kernel(
    const __bf16* __restrict__ h, const int* __restrict__ batch,
    float* __restrict__ g) {
    const int node = blockIdx.x;
    const int gi = batch[node];
    const int c = threadIdx.x;
    gatomic_add_f32(g + (size_t)gi * HID + c, (float)h[(size_t)node * HID + c]);
}

// ---- Head GEMM: z = G @ W + b (mean: blockIdx.y==0, logvar: ==1) ---------
// G:[512,128] f32, packed W:[128,64]. 4 waves cover 64 cols; 32 row blocks.
__global__ __launch_bounds__(128) void head_gemm_kernel(
    const float* __restrict__ G,
    const __bf16* __restrict__ WPm, const float* __restrict__ bm,
    const __bf16* __restrict__ WPl, const float* __restrict__ bl,
    float* __restrict__ out) {
    const __bf16* __restrict__ Wp  = blockIdx.y ? WPl : WPm;
    const float* __restrict__ bias = blockIdx.y ? bl : bm;
    float* __restrict__ o = out + (size_t)blockIdx.y * NUM_GRAPHS * LAT;

    const int lane = threadIdx.x & 31;
    const int wave = threadIdx.x >> 5;
    const int hi   = lane >> 4;
    const int l15  = lane & 15;
    const int rowBase = blockIdx.x * 16;
    const int row = rowBase + l15;
    const int col = wave * 16 + l15;

    const float* __restrict__ grow = G + (size_t)row * HID;
    const v16bf* __restrict__ bp = (const v16bf*)Wp + (size_t)(wave * 4) * 32 + lane;
    v8f acc = {};
#pragma unroll
    for (int kc = 0; kc < 4; ++kc) {
        v16bf a;
        load_a(grow, kc * 32 + hi * 8, a);
        v16bf b = bp[kc * 32];
        acc = __builtin_amdgcn_wmma_f32_16x16x32_bf16(false, a, false, b,
                                                      (short)0, acc, false, false);
    }
    const float bv = bias[col];
#pragma unroll
    for (int r = 0; r < 8; ++r) {
        const int m = rowBase + r + hi * 8;
        o[(size_t)m * LAT + col] = acc[r] + bv;
    }
}

extern "C" void kernel_launch(void* const* d_in, const int* in_sizes, int n_in,
                              void* d_out, int out_size, void* d_ws, size_t ws_size,
                              hipStream_t stream) {
    (void)in_sizes; (void)n_in; (void)out_size; (void)ws_size;
    const float* x        = (const float*)d_in[0];
    const int*   ei       = (const int*)d_in[1];
    const int*   batch    = (const int*)d_in[2];
    const float* W_in     = (const float*)d_in[3];
    const float* b_in     = (const float*)d_in[4];
    const float* W_msg    = (const float*)d_in[5];
    const float* b_msg    = (const float*)d_in[6];
    const float* W_upd    = (const float*)d_in[7];
    const float* b_upd    = (const float*)d_in[8];
    const float* W_mean   = (const float*)d_in[9];
    const float* b_mean   = (const float*)d_in[10];
    const float* W_logvar = (const float*)d_in[11];
    const float* b_logvar = (const float*)d_in[12];

    const int* src = ei;
    const int* dst = ei + N_EDGES;

    // Workspace (~103 MB): two bf16 activation buffers, one f32 agg buffer,
    // pooled graph matrix, pre-packed bf16 weights.
    const size_t NB = (size_t)N_NODES * HID;
    __bf16* P0 = (__bf16*)d_ws;
    __bf16* P1 = P0 + NB;
    float*  P2 = (float*)(P1 + NB);                  // agg (f32)
    float*  G  = P2 + NB;
    __bf16* WP = (__bf16*)(G + (size_t)NUM_GRAPHS * HID);

    const size_t NPACK128 = 16384;                   // (128/16)*2048
    const size_t NPACK64  = 8192;                    // (64/16)*2048
    __bf16* WPin   = WP;
    __bf16* WPmsg  = WP + NPACK128;                  // 3 layer matrices
    __bf16* WPupd  = WP + NPACK128 * 4;              // 3 layer matrices
    __bf16* WPmean = WP + NPACK128 * 7;
    __bf16* WPlogv = WPmean + NPACK64;

    // Pre-pack all weights into WMMA B-operand bf16 layout (0.25 MB total).
    pack_w_kernel<<<64, 256, 0, stream>>>(W_in, WPin, HID);
    for (int l = 0; l < NUM_LAYERS; ++l) {
        pack_w_kernel<<<64, 256, 0, stream>>>(W_msg + (size_t)l * HID * HID,
                                              WPmsg + (size_t)l * NPACK128, HID);
        pack_w_kernel<<<64, 256, 0, stream>>>(W_upd + (size_t)l * HID * HID,
                                              WPupd + (size_t)l * NPACK128, HID);
    }
    pack_w_kernel<<<32, 256, 0, stream>>>(W_mean,   WPmean, LAT);
    pack_w_kernel<<<32, 256, 0, stream>>>(W_logvar, WPlogv, LAT);

    const int nblk = N_NODES / 16;                   // 6250 exact: EXEC all-ones

    // h0 = relu(x @ W_in + b_in)  (f32 in, bf16 out)
    node_gemm_kernel<float, false><<<nblk, 256, 0, stream>>>(x, WPin, b_in, nullptr, P0);

    __bf16* h   = P0;
    __bf16* m   = P1;    // msg buffer, reused as h_next
    float*  agg = P2;
    for (int l = 0; l < NUM_LAYERS; ++l) {
        // msg = relu(h @ W_msg + b)   (bf16 in, bf16 out, no cvt in A path)
        node_gemm_kernel<__bf16, false><<<nblk, 256, 0, stream>>>(
            h, WPmsg + (size_t)l * NPACK128, b_msg + (size_t)l * HID, nullptr, m);
        hipMemsetAsync(agg, 0, NB * sizeof(float), stream);
        scatter_kernel<<<N_EDGES / 8, 256, 0, stream>>>(m, src, dst, agg);
        // h_next = relu(agg @ W_upd + b) + h  (f32 in, bf16 residual, bf16 out)
        node_gemm_kernel<float, true><<<nblk, 256, 0, stream>>>(
            agg, WPupd + (size_t)l * NPACK128, b_upd + (size_t)l * HID, h, m);
        __bf16* t = h; h = m; m = t;
    }

    hipMemsetAsync(G, 0, (size_t)NUM_GRAPHS * HID * sizeof(float), stream);
    pool_kernel<<<N_NODES, 128, 0, stream>>>(h, batch, G);

    head_gemm_kernel<<<dim3(NUM_GRAPHS / 16, 2), 128, 0, stream>>>(
        G, WPmean, b_mean, WPlogv, b_logvar, (float*)d_out);
}
